// GatedAttnRNN_3435973837270
// MI455X (gfx1250) — matrix operand order repaired and checked by reading, not compiled, not run on hardware
//
#include <hip/hip_runtime.h>
#include <cstdint>
#include <cstddef>

// ---------------------------------------------------------------------------
// GatedAttnRNN for MI455X (gfx1250, wave32, WMMA bf16 w/ f32 accumulate)
//
//  H=256, B=64, S=512, Q=64.  2B=128 recurrent rows (fwd + reversed passage).
//  8 persistent workgroups x 16 rows; h stays in LDS for all 512 steps.
//  uQq precomputed by a parallel WMMA GEMM; weights converted to bf16 once.
//  v2: bf16 LDS mirror of h (no f32->bf16 VALU on the recurrent path);
//      uqq_gemm A operand taken from pre-converted bf16 q2.
// ---------------------------------------------------------------------------

#define DEVI __device__ __forceinline__

typedef __attribute__((ext_vector_type(16))) __bf16 v16bf;
typedef __attribute__((ext_vector_type(8)))  float  v8f;

constexpr int H  = 256;
constexpr int H2 = 512;
constexpr int H3 = 768;
constexpr int H4 = 1024;
constexpr int Bc = 64;
constexpr int Sc = 512;
constexpr int Qc = 64;

// ----------------------------- scalar helpers ------------------------------

DEVI __bf16 f2bf(float f) {
  union { float f; unsigned u; } a; a.f = f;
  unsigned r = a.u + 0x7FFFu + ((a.u >> 16) & 1u);   // round-to-nearest-even
  union { unsigned short s; __bf16 b; } o; o.s = (unsigned short)(r >> 16);
  return o.b;
}

DEVI float bf2f(__bf16 b) {
  union { unsigned short s; __bf16 b; } i; i.b = b;
  union { unsigned u; float f; } o; o.u = ((unsigned)i.s) << 16;
  return o.f;
}

DEVI v8f vzero() {
  v8f v;
#pragma unroll
  for (int r = 0; r < 8; ++r) v[r] = 0.0f;
  return v;
}

DEVI float fast_tanh(float x) {
  x = fminf(9.0f, fmaxf(-9.0f, x));
  float e = __builtin_amdgcn_exp2f(x * 2.88539008f);          // e^(2x)
  return (e - 1.0f) * __builtin_amdgcn_rcpf(e + 1.0f);
}

DEVI float fast_sigm(float x) {
  x = fminf(30.0f, fmaxf(-30.0f, x));
  float e = __builtin_amdgcn_exp2f(-x * 1.44269504f);         // e^(-x)
  return __builtin_amdgcn_rcpf(1.0f + e);
}

// ------------------------------ WMMA helpers -------------------------------

DEVI v8f wmma_bf16(v16bf a, v16bf b, v8f c) {
  return __builtin_amdgcn_wmma_f32_16x16x32_bf16(
      /*neg_a=*/false, a, /*neg_b=*/false, b,
      /*c_mod=*/(short)0, c, /*reuse_a=*/false, /*reuse_b=*/false);
}

// A fragment: 16x32 bf16.  Lane L: M=L&15, K-base=(L>>4)*16;
// element e -> K = K-base + e.
DEVI v16bf ld_a_bf16(const __bf16* base, int ld) {
  int L = threadIdx.x & 31;
  const __bf16* p = base + (size_t)(L & 15) * ld + ((L >> 4) << 4);
  v16bf v;
#pragma unroll
  for (int e = 0; e < 16; ++e) v[e] = p[e];
  return v;
}

// B fragment: 32x16 bf16 where B[k][n] = W[n][k] (W row-major NxK).
// Lane L: N=L&15, K-base=(L>>4)*16, 16 consecutive k -> two b128 loads.
DEVI v16bf ld_b_bf16(const __bf16* base, int ldk) {
  int L = threadIdx.x & 31;
  const __bf16* p = base + (size_t)(L & 15) * ldk + ((L >> 4) << 4);
  union { uint4 q[2]; v16bf v; } u;
  u.q[0] = *reinterpret_cast<const uint4*>(p);
  u.q[1] = *reinterpret_cast<const uint4*>(p + 8);
  return u.v;
}

// C/D fragment store: element r -> M=(L>>4)*8+r, N=L&15.
DEVI void st_c_f32(float* base, int ld, v8f c) {
  int L = threadIdx.x & 31;
  float* p = base + (size_t)((L >> 4) << 3) * ld + (L & 15);
#pragma unroll
  for (int r = 0; r < 8; ++r) p[(size_t)r * ld] = c[r];
}

// ------------------------- weight conversion kernel ------------------------

__global__ __launch_bounds__(256) void cvt_f32_to_bf16(const float* __restrict__ src,
                                                       __bf16* __restrict__ dst, int n) {
  int i = blockIdx.x * 256 + threadIdx.x;
  if (i < n) dst[i] = f2bf(src[i]);
}

// -------------------- uQq = q2 @ WuQ^T  (4096 x 512 x 256) -----------------
// grid: 512 blocks x 256 threads (8 waves); one 16x16 tile per wave.
// A operand comes from the pre-converted bf16 q2 copy (no per-frag cvt VALU).

__global__ __launch_bounds__(256) void uqq_gemm(const __bf16* __restrict__ q2b,
                                                const __bf16* __restrict__ WuQb,
                                                float* __restrict__ uQq) {
  int wave = threadIdx.x >> 5;
  int tile = blockIdx.x * 8 + wave;       // 4096 tiles: 256 M-tiles x 16 N-tiles
  int mt = tile >> 4;
  int nt = tile & 15;
  const __bf16* A0 = q2b  + (size_t)(mt * 16) * H2;
  const __bf16* B0 = WuQb + (size_t)(nt * 16) * H2;
  v8f acc = vzero();
#pragma unroll
  for (int ks = 0; ks < 16; ++ks) {
    v16bf a = ld_a_bf16(A0 + ks * 32, H2);
    v16bf b = ld_b_bf16(B0 + ks * 32, H2);
    acc = wmma_bf16(a, b, acc);
  }
  int L = threadIdx.x & 31;
  float* C = uQq + (size_t)(mt * 16) * H + nt * 16;
  float* p = C + (size_t)((L >> 4) << 3) * H + (L & 15);
#pragma unroll
  for (int r = 0; r < 8; ++r) p[(size_t)r * H] = acc[r];
}

// ------------------------------ the scan kernel ----------------------------
// 8 blocks x 512 threads (16 waves).  Block bx owns rows [16*bx, 16*bx+16).
// Row r: d = r>>6 (0 fwd / 1 reversed), b = r&63.

__global__ __launch_bounds__(512, 1) void scan_kernel(
    const float* __restrict__ pe,        // p_enc (B,S,2H)
    const int*   __restrict__ qlen,      // (B,)
    const int*   __restrict__ plen,      // (B,)
    const float* __restrict__ wvt,       // WvT (H,)
    const float* __restrict__ bih,       // (3H,)
    const float* __restrict__ bhh,       // (3H,)
    const __bf16* __restrict__ WuPb,     // (H,2H)
    const __bf16* __restrict__ WvPb,     // (H,H)
    const __bf16* __restrict__ Wgb,      // (4H,4H)
    const __bf16* __restrict__ Wihb,     // (3H,4H)
    const __bf16* __restrict__ Whhb,     // (3H,H)
    const __bf16* __restrict__ q2b,      // q_enc bf16 (B,Q,2H)
    const float* __restrict__ uQq,       // (B,Q,H)
    float* __restrict__ out)             // (B,S,2H)
{
  __shared__ float  sh_h[16 * H];        // hidden state fp32, persistent
  __shared__ __bf16 sh_hb[16 * H];       // hidden state bf16 mirror (WMMA A)
  __shared__ __bf16 sh_upb[16 * H2];     // up_t bf16
  __shared__ float  sh_cvec[16 * H];     // up@WuP^T + h@WvP^T
  __shared__ float  sh_sc[16 * Qc];      // scores -> attention weights
  __shared__ float  sh_ctf[16 * H2];     // context fp32
  __shared__ __bf16 sh_ctb[16 * H2];     // context bf16
  __shared__ float  sh_gacc[16 * H4];    // gate preact; reused as gi (16x768)
  __shared__ __bf16 sh_xb[16 * H4];      // x_in bf16
  __shared__ float  sh_ghh[16 * H3];     // h@Whh^T

  const int tid  = threadIdx.x;
  const int w    = tid >> 5;             // wave == local row
  const int lane = tid & 31;
  const int rG   = blockIdx.x * 16 + w;
  const int d    = rG >> 6;
  const int b    = rG & 63;
  const int pl   = plen[b];
  const int ql   = qlen[b];

  // zero hidden state (both copies)
  for (int i = tid; i < 16 * H; i += 512) { sh_h[i] = 0.0f; sh_hb[i] = f2bf(0.0f); }

  // hoist per-lane WvT chunk (h = lane*8 .. +8)
  float wv[8];
#pragma unroll
  for (int i = 0; i < 8; ++i) wv[i] = wvt[lane * 8 + i];

  const float* uqq_base = uQq + (size_t)b * Qc * H + lane * 8;

  __syncthreads();

  for (int t = 0; t < Sc; ++t) {
    // ---- P0: fetch up_t row (with in-flight sequence reversal), to bf16 ----
    const int tsrc = d ? ((t < pl) ? (pl - 1 - t) : t) : t;
    const float* uprow = pe + ((size_t)b * Sc + tsrc) * H2;
    for (int j = lane; j < H2; j += 32) sh_upb[w * H2 + j] = f2bf(uprow[j]);
    __syncthreads();

    // ---- P1: cvec = up@WuP^T + h@WvP^T   (wave w -> N-tile w) --------------
    {
      const int n0 = w * 16;
      v8f acc = vzero();
#pragma unroll
      for (int ks = 0; ks < 16; ++ks) {
        v16bf a = ld_a_bf16(sh_upb + ks * 32, H2);
        v16bf bm = ld_b_bf16(WuPb + (size_t)n0 * H2 + ks * 32, H2);
        acc = wmma_bf16(a, bm, acc);
      }
#pragma unroll
      for (int ks = 0; ks < 8; ++ks) {
        v16bf a = ld_a_bf16(sh_hb + ks * 32, H);
        v16bf bm = ld_b_bf16(WvPb + (size_t)n0 * H + ks * 32, H);
        acc = wmma_bf16(a, bm, acc);
      }
      st_c_f32(sh_cvec + n0, H, acc);
    }
    __syncthreads();

    // ---- P2: scores s[w][q] = sum_h tanh(uQq + cvec) * WvT  ---------------
    {
      float cv[8];
      const float4* c4 = reinterpret_cast<const float4*>(sh_cvec + w * H + lane * 8);
      float4 c0 = c4[0], c1 = c4[1];
      cv[0]=c0.x; cv[1]=c0.y; cv[2]=c0.z; cv[3]=c0.w;
      cv[4]=c1.x; cv[5]=c1.y; cv[6]=c1.z; cv[7]=c1.w;
      for (int q = 0; q < Qc; ++q) {
        const float4* u4 = reinterpret_cast<const float4*>(uqq_base + (size_t)q * H);
        float4 u0 = u4[0], u1 = u4[1];
        float s = fast_tanh(u0.x + cv[0]) * wv[0];
        s += fast_tanh(u0.y + cv[1]) * wv[1];
        s += fast_tanh(u0.z + cv[2]) * wv[2];
        s += fast_tanh(u0.w + cv[3]) * wv[3];
        s += fast_tanh(u1.x + cv[4]) * wv[4];
        s += fast_tanh(u1.y + cv[5]) * wv[5];
        s += fast_tanh(u1.z + cv[6]) * wv[6];
        s += fast_tanh(u1.w + cv[7]) * wv[7];
#pragma unroll
        for (int o = 16; o > 0; o >>= 1) s += __shfl_down(s, o);
        if (lane == 0) sh_sc[w * Qc + q] = s;
      }
    }
    __syncthreads();

    // ---- P3: masked softmax over Q (row w) --------------------------------
    {
      float s0 = (lane      < ql) ? sh_sc[w * Qc + lane]      : -1e30f;
      float s1 = (lane + 32 < ql) ? sh_sc[w * Qc + 32 + lane] : -1e30f;
      float m = fmaxf(s0, s1);
#pragma unroll
      for (int o = 16; o > 0; o >>= 1) m = fmaxf(m, __shfl_xor(m, o));
      float e0 = __builtin_amdgcn_exp2f((s0 - m) * 1.44269504f);
      float e1 = __builtin_amdgcn_exp2f((s1 - m) * 1.44269504f);
      float sum = e0 + e1;
#pragma unroll
      for (int o = 16; o > 0; o >>= 1) sum += __shfl_xor(sum, o);
      float rs = __builtin_amdgcn_rcpf(sum);
      sh_sc[w * Qc + lane]      = e0 * rs;
      sh_sc[w * Qc + 32 + lane] = e1 * rs;
    }
    __syncthreads();

    // ---- P4: ct[w][:] = at @ q2[b]  (per-row B -> VALU, coalesced bf16) ---
    {
      float acc[16];
#pragma unroll
      for (int i = 0; i < 16; ++i) acc[i] = 0.0f;
      const __bf16* qb = q2b + (size_t)b * Qc * H2 + lane * 16;
      for (int q = 0; q < Qc; ++q) {
        float a = sh_sc[w * Qc + q];
        union { uint4 u[2]; __bf16 h[16]; } uu;
        const uint4* p4 = reinterpret_cast<const uint4*>(qb + (size_t)q * H2);
        uu.u[0] = p4[0]; uu.u[1] = p4[1];
#pragma unroll
        for (int i = 0; i < 16; ++i) acc[i] += a * bf2f(uu.h[i]);
      }
#pragma unroll
      for (int i = 0; i < 16; ++i) {
        sh_ctf[w * H2 + lane * 16 + i] = acc[i];
        sh_ctb[w * H2 + lane * 16 + i] = f2bf(acc[i]);
      }
    }
    __syncthreads();

    // ---- P5: gate preact = [up,ct] @ Wg^T  (N=1024, 4 N-tiles per wave) ---
    __builtin_prefetch(Wihb + (size_t)(w * 3 * 16) * H4, 0, 2);
    for (int s5 = 0; s5 < 4; ++s5) {
      const int n0 = (w * 4 + s5) * 16;
      v8f acc = vzero();
#pragma unroll
      for (int ks = 0; ks < 16; ++ks) {
        v16bf a = ld_a_bf16(sh_upb + ks * 32, H2);
        v16bf bm = ld_b_bf16(Wgb + (size_t)n0 * H4 + ks * 32, H4);
        acc = wmma_bf16(a, bm, acc);
      }
#pragma unroll
      for (int ks = 0; ks < 16; ++ks) {
        v16bf a = ld_a_bf16(sh_ctb + ks * 32, H2);
        v16bf bm = ld_b_bf16(Wgb + (size_t)n0 * H4 + H2 + ks * 32, H4);
        acc = wmma_bf16(a, bm, acc);
      }
      st_c_f32(sh_gacc + n0, H4, acc);
    }
    __syncthreads();

    // ---- P6: x_in = sigmoid(gate) * [up_f32, ct_f32]  -> bf16 -------------
    {
      for (int jj = 0; jj < 32; ++jj) {
        int j = lane * 32 + jj;
        float g = sh_gacc[w * H4 + j];
        float upc = (j < H2) ? uprow[j] : sh_ctf[w * H2 + (j - H2)];
        sh_xb[w * H4 + j] = f2bf(fast_sigm(g) * upc);
      }
    }
    __syncthreads();

    // ---- P7a: gh = h @ Whh^T  (N=768, 3 tiles per wave) -------------------
    for (int s7 = 0; s7 < 3; ++s7) {
      const int n0 = (w * 3 + s7) * 16;
      v8f acc = vzero();
#pragma unroll
      for (int ks = 0; ks < 8; ++ks) {
        v16bf a = ld_a_bf16(sh_hb + ks * 32, H);
        v16bf bm = ld_b_bf16(Whhb + (size_t)n0 * H + ks * 32, H);
        acc = wmma_bf16(a, bm, acc);
      }
      st_c_f32(sh_ghh + n0, H3, acc);
    }
    __syncthreads();

    // ---- P7b: gi = x_in @ Wih^T  (reuse sh_gacc as 16x768 buffer) ---------
    for (int s7 = 0; s7 < 3; ++s7) {
      const int n0 = (w * 3 + s7) * 16;
      v8f acc = vzero();
#pragma unroll
      for (int ks = 0; ks < 32; ++ks) {
        v16bf a = ld_a_bf16(sh_xb + ks * 32, H4);
        v16bf bm = ld_b_bf16(Wihb + (size_t)n0 * H4 + ks * 32, H4);
        acc = wmma_bf16(a, bm, acc);
      }
      st_c_f32(sh_gacc + n0, H3, acc);
    }
    __syncthreads();

    // ---- P8: GRU combine + mask + h update (f32 + bf16 mirror) + output ---
    {
      const bool act = (t < pl);
#pragma unroll
      for (int mm = 0; mm < 8; ++mm) {
        int m = lane * 8 + mm;
        float ir  = sh_gacc[w * H3 + m]           + bih[m];
        float hr  = sh_ghh [w * H3 + m]           + bhh[m];
        float iz  = sh_gacc[w * H3 + H + m]       + bih[H + m];
        float hz  = sh_ghh [w * H3 + H + m]       + bhh[H + m];
        float inn = sh_gacc[w * H3 + 2 * H + m]   + bih[2 * H + m];
        float hn  = sh_ghh [w * H3 + 2 * H + m]   + bhh[2 * H + m];
        float rg = fast_sigm(ir + hr);
        float z  = fast_sigm(iz + hz);
        float n  = fast_tanh(inn + rg * hn);
        float hp = sh_h[w * H + m];
        float hnew = (1.0f - z) * n + z * hp;
        float hout = act ? hnew : 0.0f;
        sh_h[w * H + m]  = hout;
        sh_hb[w * H + m] = f2bf(hout);
        out[((size_t)b * Sc + t) * H2 + (size_t)d * H + m] = hout;
      }
    }
    __syncthreads();
  }
}

// ------------------------------- launcher ----------------------------------

extern "C" void kernel_launch(void* const* d_in, const int* in_sizes, int n_in,
                              void* d_out, int out_size, void* d_ws, size_t ws_size,
                              hipStream_t stream) {
  (void)in_sizes; (void)n_in; (void)out_size; (void)ws_size;

  const float* q_enc = (const float*)d_in[0];
  const float* p_enc = (const float*)d_in[1];
  const int*   q_len = (const int*)d_in[2];
  const int*   p_len = (const int*)d_in[3];
  const float* WuQ   = (const float*)d_in[4];
  const float* WuP   = (const float*)d_in[5];
  const float* WvP   = (const float*)d_in[6];
  const float* WvT   = (const float*)d_in[7];
  const float* Wg    = (const float*)d_in[8];
  const float* Wih   = (const float*)d_in[9];
  const float* Whh   = (const float*)d_in[10];
  const float* bih   = (const float*)d_in[11];
  const float* bhh   = (const float*)d_in[12];
  float* out = (float*)d_out;

  char* wsp = (char*)d_ws;
  size_t off = 0;
  auto take = [&](size_t bytes) -> char* {
    char* p = wsp + off;
    off = (off + bytes + 255) & ~(size_t)255;
    return p;
  };

  __bf16* WuQb = (__bf16*)take((size_t)H  * H2 * 2);
  __bf16* WuPb = (__bf16*)take((size_t)H  * H2 * 2);
  __bf16* WvPb = (__bf16*)take((size_t)H  * H  * 2);
  __bf16* Wgb  = (__bf16*)take((size_t)H4 * H4 * 2);
  __bf16* Wihb = (__bf16*)take((size_t)H3 * H4 * 2);
  __bf16* Whhb = (__bf16*)take((size_t)H3 * H  * 2);
  __bf16* q2b  = (__bf16*)take((size_t)Bc * Qc * H2 * 2);
  float*  uQq  = (float*) take((size_t)Bc * Qc * H * 4);

  auto cvt = [&](const float* s, __bf16* dst, int n) {
    cvt_f32_to_bf16<<<(n + 255) / 256, 256, 0, stream>>>(s, dst, n);
  };
  cvt(WuQ,   WuQb, H * H2);
  cvt(WuP,   WuPb, H * H2);
  cvt(WvP,   WvPb, H * H);
  cvt(Wg,    Wgb,  H4 * H4);
  cvt(Wih,   Wihb, H3 * H4);
  cvt(Whh,   Whhb, H3 * H);
  cvt(q_enc, q2b,  Bc * Qc * H2);

  // uQq = q2 @ WuQ^T : 4096x512x256, 4096 wave-tiles, 8 waves per block
  uqq_gemm<<<512, 256, 0, stream>>>(q2b, WuQb, uQq);

  // persistent sequential scan: 8 blocks x 16 waves
  scan_kernel<<<8, 512, 0, stream>>>(p_enc, q_len, p_len, WvT, bih, bhh,
                                     WuPb, WvPb, Wgb, Wihb, Whhb, q2b, uQq, out);
}
